// DGCNN_semseg_80917183857082
// MI455X (gfx1250) — matrix operand order repaired and checked by reading, not compiled, not run on hardware
//
#include <hip/hip_runtime.h>
#include <hip/hip_bf16.h>

// DGCNN semseg forward for MI455X (gfx1250, wave32).
// All GEMM-shaped work (knn inner products, edge convs, cat GEMM) runs on
// v_wmma_f32_16x16x32_bf16. Requires ws_size >= ~210 MB.

#define BN_EPS 1e-5f
#define TOPK 20
#define NEG_INF (-3.402823466e38f)

typedef __attribute__((ext_vector_type(16))) __bf16 bf16x16;
typedef __attribute__((ext_vector_type(8)))  float  f32x8;
typedef __attribute__((ext_vector_type(4)))  unsigned int u32x4;
typedef __attribute__((ext_vector_type(8)))  unsigned int u32x8;

__device__ __forceinline__ f32x8 wmma_bf16(bf16x16 a, bf16x16 b, f32x8 c) {
  return __builtin_amdgcn_wmma_f32_16x16x32_bf16(false, a, false, b, (short)0, c,
                                                 false, false);
}

__device__ __forceinline__ bf16x16 frag_from(const __bf16* p0, const __bf16* p1) {
  u32x4 lo = *(const u32x4*)p0;   // 8 bf16
  u32x4 hi = *(const u32x4*)p1;   // 8 bf16
  u32x8 t;
  t[0] = lo[0]; t[1] = lo[1]; t[2] = lo[2]; t[3] = lo[3];
  t[4] = hi[0]; t[5] = hi[1]; t[6] = hi[2]; t[7] = hi[3];
  return __builtin_bit_cast(bf16x16, t);
}

// A-matrix fragment (16x32 bf16). `row` = pointer to this lane's matrix row
// (lane&15). lanes 0-15 hold K = kb+0..7 and kb+16..23; lanes 16-31 hold
// K = kb+8..15 and kb+24..31 (ISA 7.12.2).
__device__ __forceinline__ bf16x16 ld_afrag(const __bf16* row, int kb, int hf) {
  return frag_from(row + kb + 8 * hf, row + kb + 16 + 8 * hf);
}
// B-matrix fragment (32x16 bf16). `col` = pointer to this lane's column data
// (contraction-contiguous). lanes 0-15: K=kb+0..7, kb+16..23; lanes 16-31: kb+8..15, kb+24..31.
__device__ __forceinline__ bf16x16 ld_bfrag(const __bf16* col, int kb, int hf) {
  return frag_from(col + kb + 8 * hf, col + kb + 16 + 8 * hf);
}

__device__ __forceinline__ float lrelu(float v) { return v >= 0.f ? v : 0.2f * v; }

// ---------------------------------------------------------------- prep kernels
__global__ void transpose_x_kernel(const float* __restrict__ x, float* __restrict__ p,
                                   int B, int N) {
  int t = blockIdx.x * blockDim.x + threadIdx.x;
  if (t >= B * N * 3) return;
  int b = t / (N * 3);
  int rr = t - b * (N * 3);
  int n = rr / 3, c = rr - n * 3;
  p[t] = x[((size_t)b * 3 + c) * N + n];
}

__global__ void wconv_kernel(const float* __restrict__ w, __bf16* __restrict__ wo,
                             int O, int Cin, int Cpad) {
  int t = blockIdx.x * blockDim.x + threadIdx.x;
  if (t >= O * Cpad) return;
  int o = t / Cpad, c = t - o * Cpad;
  wo[t] = (__bf16)((c < Cin) ? w[(size_t)o * Cin + c] : 0.f);
}

__global__ void rownorm_kernel(const float* __restrict__ X, float* __restrict__ xx,
                               int rows, int C) {
  int r = blockIdx.x * blockDim.x + threadIdx.x;
  if (r >= rows) return;
  const float* p = X + (size_t)r * C;
  float s = 0.f;
  for (int c = 0; c < C; ++c) s += p[c] * p[c];
  xx[r] = s;
}

// ------------------------------------------------------------- knn: distances
// Stage 1: C=3, direct negative squared distance (one batch's N x N matrix).
__global__ void dist3_kernel(const float* __restrict__ p, float* __restrict__ dist,
                             int N) {
  size_t t = (size_t)blockIdx.x * blockDim.x + threadIdx.x;
  if (t >= (size_t)N * N) return;
  int n = (int)(t / N), m = (int)(t - (size_t)n * N);
  float dx = p[n * 3 + 0] - p[m * 3 + 0];
  float dy = p[n * 3 + 1] - p[m * 3 + 1];
  float dz = p[n * 3 + 2] - p[m * 3 + 2];
  dist[t] = -(dx * dx + dy * dy + dz * dz);
}

// Stages 2/3: C=64. neg_dist = 2*X@X^T - xx_n - xx_m via WMMA bf16.
// One wave computes a 16(n) x 64(m) tile. 8 waves / block.
__global__ void knn_dist_wmma_kernel(const __bf16* __restrict__ X,
                                     const float* __restrict__ xx,
                                     float* __restrict__ dist, int N, int C) {
  int lane = threadIdx.x & 31, wave = threadIdx.x >> 5;
  int hf = lane >> 4, l16 = lane & 15;
  int waveId = blockIdx.x * 8 + wave;
  int mBlocks = N >> 6;
  int nt = waveId / mBlocks;
  int mb = waveId - nt * mBlocks;
  const __bf16* arow = X + (size_t)(nt * 16 + l16) * C;
  f32x8 z = {0.f, 0.f, 0.f, 0.f, 0.f, 0.f, 0.f, 0.f};
  f32x8 acc[4] = {z, z, z, z};
  for (int kb = 0; kb < C; kb += 32) {
    bf16x16 a = ld_afrag(arow, kb, hf);
#pragma unroll
    for (int t = 0; t < 4; ++t) {
      const __bf16* bcol = X + (size_t)(mb * 64 + t * 16 + l16) * C;
      acc[t] = wmma_bf16(a, ld_bfrag(bcol, kb, hf), acc[t]);
    }
  }
  int n0 = nt * 16;
#pragma unroll
  for (int t = 0; t < 4; ++t) {
    int m = mb * 64 + t * 16 + l16;
    float xm = xx[m];
#pragma unroll
    for (int v = 0; v < 8; ++v) {
      int n = n0 + v + 8 * hf;
      dist[(size_t)n * N + m] = 2.0f * acc[t][v] - xx[n] - xm;
    }
  }
}

// ----------------------------------------------------------------- knn: top-k
// One wave per row. Per-lane register-resident sorted top-20, then 32-way
// merge through LDS with shfl_xor max reductions (tie -> lower index, stable).
__global__ void topk_kernel(const float* __restrict__ dist, int* __restrict__ out,
                            int N) {
  __shared__ float sv[8][32 * TOPK];
  __shared__ int si[8][32 * TOPK];
  int lane = threadIdx.x & 31, wave = threadIdx.x >> 5;
  int n = blockIdx.x * 8 + wave;
  const float* row = dist + (size_t)n * N;
  float tv[TOPK];
  int ti[TOPK];
#pragma unroll
  for (int j = 0; j < TOPK; ++j) { tv[j] = NEG_INF; ti[j] = 0x7fffffff; }
  for (int m = lane; m < N; m += 32) {
    float v = row[m];
    if (v > tv[TOPK - 1]) {
      tv[TOPK - 1] = v; ti[TOPK - 1] = m;
#pragma unroll
      for (int j = TOPK - 1; j > 0; --j) {
        if (tv[j] > tv[j - 1]) {
          float fv = tv[j]; tv[j] = tv[j - 1]; tv[j - 1] = fv;
          int fi = ti[j]; ti[j] = ti[j - 1]; ti[j - 1] = fi;
        }
      }
    }
  }
#pragma unroll
  for (int j = 0; j < TOPK; ++j) {
    sv[wave][lane * TOPK + j] = tv[j];
    si[wave][lane * TOPK + j] = ti[j];
  }
  int head = 0;
  for (int r = 0; r < TOPK; ++r) {
    float v = (head < TOPK) ? sv[wave][lane * TOPK + head] : NEG_INF;
    int ii = (head < TOPK) ? si[wave][lane * TOPK + head] : 0x7fffffff;
    int wl = lane;
#pragma unroll
    for (int off = 16; off > 0; off >>= 1) {
      float ov = __shfl_xor(v, off);
      int oi = __shfl_xor(ii, off);
      int ol = __shfl_xor(wl, off);
      if (ov > v || (ov == v && oi < ii)) { v = ov; ii = oi; wl = ol; }
    }
    if (lane == wl) head++;
    if (lane == 0) out[(size_t)n * TOPK + r] = ii;
  }
}

// ------------------------------------------------- edge-conv (gather + WMMA)
// y[row,o] = sum_c e[row,c] * W[o,c], e = [nb-ctr | ctr | 0-pad], rows=B*N*Kn.
// Wave builds its 16-row x Ktot A-tile in LDS (same-wave DS ordering => no
// barrier), then 16x64 WMMA. Ktot in {32,128}, O=64.
__global__ void edgeconv_wmma_kernel(const float* __restrict__ feat,
                                     const int* __restrict__ nbr,
                                     const __bf16* __restrict__ W,
                                     float* __restrict__ y, int Cin, int Ktot,
                                     int N, int Kn, int O) {
  __shared__ __align__(16) __bf16 sA[8][16 * 128];
  int lane = threadIdx.x & 31, wave = threadIdx.x >> 5;
  int hf = lane >> 4, l16 = lane & 15;
  long long rowbase = ((long long)blockIdx.x * 8 + wave) * 16;
  __bf16* A = sA[wave];
  int tot = 16 * Ktot;
  for (int e = lane; e < tot; e += 32) {
    int r = e / Ktot, c = e - r * Ktot;
    long long gr = rowbase + r;
    int b = (int)(gr / ((long long)N * Kn));
    long long rem = gr - (long long)b * N * Kn;
    int n = (int)(rem / Kn);
    float val = 0.f;
    if (c < 2 * Cin) {
      int cc = (c < Cin) ? c : (c - Cin);
      float ctr = feat[((size_t)b * N + n) * Cin + cc];
      if (c < Cin) {
        int j = nbr[gr];
        val = feat[((size_t)b * N + j) * Cin + c] - ctr;
      } else {
        val = ctr;
      }
    }
    A[e] = (__bf16)val;
  }
  f32x8 z = {0.f, 0.f, 0.f, 0.f, 0.f, 0.f, 0.f, 0.f};
  f32x8 acc[4] = {z, z, z, z};
  const __bf16* arow = A + l16 * Ktot;
  for (int kb = 0; kb < Ktot; kb += 32) {
    bf16x16 a = ld_afrag(arow, kb, hf);
#pragma unroll
    for (int t = 0; t < 4; ++t) {
      const __bf16* wcol = W + (size_t)(t * 16 + l16) * Ktot;
      acc[t] = wmma_bf16(a, ld_bfrag(wcol, kb, hf), acc[t]);
    }
  }
#pragma unroll
  for (int t = 0; t < 4; ++t)
#pragma unroll
    for (int v = 0; v < 8; ++v)
      y[(size_t)(rowbase + v + 8 * hf) * O + t * 16 + l16] = acc[t][v];
}

// ------------------------------------------ plain GEMM: y = A(bf16) @ W^T(bf16)
// grid.x covers row tiles (128 rows/block), grid.y covers 64-col blocks of O.
__global__ void gemm_wmma_kernel(const __bf16* __restrict__ A,
                                 const __bf16* __restrict__ W,
                                 float* __restrict__ y, int Ktot, int O) {
  int lane = threadIdx.x & 31, wave = threadIdx.x >> 5;
  int hf = lane >> 4, l16 = lane & 15;
  long long rowbase = ((long long)blockIdx.x * 8 + wave) * 16;
  int ob = blockIdx.y * 64;
  const __bf16* arow = A + (size_t)(rowbase + l16) * Ktot;
  f32x8 z = {0.f, 0.f, 0.f, 0.f, 0.f, 0.f, 0.f, 0.f};
  f32x8 acc[4] = {z, z, z, z};
  for (int kb = 0; kb < Ktot; kb += 32) {
    bf16x16 a = ld_afrag(arow, kb, hf);
#pragma unroll
    for (int t = 0; t < 4; ++t) {
      const __bf16* wcol = W + (size_t)(ob + t * 16 + l16) * Ktot;
      acc[t] = wmma_bf16(a, ld_bfrag(wcol, kb, hf), acc[t]);
    }
  }
#pragma unroll
  for (int t = 0; t < 4; ++t)
#pragma unroll
    for (int v = 0; v < 8; ++v)
      y[(size_t)(rowbase + v + 8 * hf) * O + ob + t * 16 + l16] = acc[t][v];
}

// --------------------------------------------- batchnorm stats (deterministic)
// Slice-parallel partial sums (fixed traversal order), then sequential per-channel sum over slices.
__global__ void bn_stats_slice_kernel(const float* __restrict__ y, long long rows,
                                      int C, int S, float* __restrict__ partial) {
  int t = blockIdx.x * blockDim.x + threadIdx.x;
  if (t >= C * S) return;
  int c = t % C, s = t / C;
  long long r0 = (long long)s * rows / S;
  long long r1 = (long long)(s + 1) * rows / S;
  float sm = 0.f, sq = 0.f;
  for (long long r = r0; r < r1; ++r) {
    float v = y[r * C + c];
    sm += v;
    sq += v * v;
  }
  partial[(size_t)s * 2 * C + c] = sm;
  partial[(size_t)s * 2 * C + C + c] = sq;
}

__global__ void bn_stats_reduce_kernel(const float* __restrict__ partial, int S,
                                       int C, float* __restrict__ stat) {
  int t = blockIdx.x * blockDim.x + threadIdx.x;
  if (t >= 2 * C) return;
  float s = 0.f;
  for (int i = 0; i < S; ++i) s += partial[(size_t)i * 2 * C + t];
  stat[t] = s;
}

__global__ void bn_finalize_kernel(const float* __restrict__ stat,
                                   const float* __restrict__ gamma,
                                   const float* __restrict__ beta,
                                   float* __restrict__ ss, int C, float invn) {
  int c = blockIdx.x * blockDim.x + threadIdx.x;
  if (c >= C) return;
  float m = stat[c] * invn;
  float v = stat[C + c] * invn - m * m;
  float sc = gamma[c] * rsqrtf(v + BN_EPS);
  ss[c] = sc;
  ss[C + c] = beta[c] - m * sc;
}

__global__ void bn_act_bf16_kernel(const float* __restrict__ y,
                                   const float* __restrict__ ss,
                                   __bf16* __restrict__ h, long long total, int C) {
  long long e = (long long)blockIdx.x * blockDim.x + threadIdx.x;
  if (e >= total) return;
  int c = (int)(e % C);
  h[e] = (__bf16)lrelu(y[e] * ss[c] + ss[C + c]);
}

// BN + LReLU + max over k; writes f32 feature, bf16 feature, and cat slice.
__global__ void bn_act_maxk_kernel(const float* __restrict__ y,
                                   const float* __restrict__ ss,
                                   float* __restrict__ xf, __bf16* __restrict__ xb,
                                   __bf16* __restrict__ cat, int catld, int catoff,
                                   long long NR, int Kn, int C) {
  long long t = (long long)blockIdx.x * blockDim.x + threadIdx.x;
  if (t >= NR * C) return;
  int c = (int)(t % C);
  long long r = t / C;
  float sc = ss[c], sh = ss[C + c];
  const float* yp = y + (size_t)r * Kn * C + c;
  float mx = NEG_INF;
  for (int k = 0; k < Kn; ++k) mx = fmaxf(mx, lrelu(yp[(size_t)k * C] * sc + sh));
  xf[t] = mx;
  xb[t] = (__bf16)mx;
  cat[r * catld + catoff + c] = (__bf16)mx;
}

// BN + LReLU + max over N (global pool). blockDim = C = 1024, blockIdx = b.
__global__ void global_maxn_kernel(const float* __restrict__ y6,
                                   const float* __restrict__ ss,
                                   float* __restrict__ g, int N, int C) {
  int o = threadIdx.x, b = blockIdx.x;
  float sc = ss[o], sh = ss[C + o];
  const float* p = y6 + (size_t)b * N * C + o;
  float mx = NEG_INF;
  for (int n = 0; n < N; ++n) mx = fmaxf(mx, lrelu(p[(size_t)n * C] * sc + sh));
  g[b * C + o] = mx;
}

__global__ void final_linear_kernel(const float* __restrict__ g,
                                    const float* __restrict__ wl,
                                    float* __restrict__ t, int B, int Cin,
                                    int Cout) {
  int i = blockIdx.x * blockDim.x + threadIdx.x;
  if (i >= B * Cout) return;
  int b = i / Cout, o = i - b * Cout;
  const float* gp = g + (size_t)b * Cin;
  const float* wp = wl + (size_t)o * Cin;
  float s = 0.f;
  for (int c = 0; c < Cin; ++c) s += gp[c] * wp[c];
  t[i] = s;
}

__global__ void final_bn_kernel(const float* __restrict__ t,
                                const float* __restrict__ g7,
                                const float* __restrict__ b7,
                                float* __restrict__ out, int B, int C) {
  int o = blockIdx.x * blockDim.x + threadIdx.x;
  if (o >= C) return;
  float m = 0.f;
  for (int b = 0; b < B; ++b) m += t[b * C + o];
  m *= (1.f / B);
  float v = 0.f;
  for (int b = 0; b < B; ++b) {
    float d = t[b * C + o] - m;
    v += d * d;
  }
  v *= (1.f / B);
  float sc = rsqrtf(v + BN_EPS) * g7[o];
  for (int b = 0; b < B; ++b) out[b * C + o] = (t[b * C + o] - m) * sc + b7[o];
}

// ================================================================== launcher
extern "C" void kernel_launch(void* const* d_in, const int* in_sizes, int n_in,
                              void* d_out, int out_size, void* d_ws, size_t ws_size,
                              hipStream_t stream) {
  (void)in_sizes; (void)n_in; (void)out_size; (void)ws_size;
  const float* x  = (const float*)d_in[0];
  const float* w1 = (const float*)d_in[1];
  const float* w2 = (const float*)d_in[2];
  const float* w3 = (const float*)d_in[3];
  const float* w4 = (const float*)d_in[4];
  const float* w5 = (const float*)d_in[5];
  const float* w6 = (const float*)d_in[6];
  const float* wl = (const float*)d_in[7];
  const float* g1 = (const float*)d_in[8];  const float* b1 = (const float*)d_in[9];
  const float* g2 = (const float*)d_in[10]; const float* b2 = (const float*)d_in[11];
  const float* g3 = (const float*)d_in[12]; const float* b3 = (const float*)d_in[13];
  const float* g4 = (const float*)d_in[14]; const float* b4 = (const float*)d_in[15];
  const float* g5 = (const float*)d_in[16]; const float* b5 = (const float*)d_in[17];
  const float* g6 = (const float*)d_in[18]; const float* b6 = (const float*)d_in[19];
  const float* g7 = (const float*)d_in[20]; const float* b7 = (const float*)d_in[21];

  constexpr int B = 4, N = 4096, Kn = TOPK;
  constexpr long long R = (long long)B * N * Kn;  // 327680 rows for edge convs
  constexpr long long NR = (long long)B * N;      // 16384 points

  char* ws = (char*)d_ws;
  size_t off = 0;
  auto alloc = [&](size_t bytes) -> void* {
    void* p = ws + off;
    off = (off + bytes + 255) & ~(size_t)255;
    return p;
  };
  float*   p    = (float*)alloc((size_t)B * N * 3 * 4);
  float*   dist = (float*)alloc((size_t)N * N * 4);         // reused; aliases y6
  int*     nbr  = (int*)alloc((size_t)R * 4);
  float*   ybuf = (float*)alloc((size_t)R * 64 * 4);
  __bf16*  hbf  = (__bf16*)alloc((size_t)R * 64 * 2);
  float*   x1f  = (float*)alloc((size_t)NR * 64 * 4);
  float*   x2f  = (float*)alloc((size_t)NR * 64 * 4);
  __bf16*  x1b  = (__bf16*)alloc((size_t)NR * 64 * 2);
  __bf16*  x2b  = (__bf16*)alloc((size_t)NR * 64 * 2);
  __bf16*  catb = (__bf16*)alloc((size_t)NR * 192 * 2);
  float*   xx   = (float*)alloc((size_t)NR * 4);
  __bf16*  w1b  = (__bf16*)alloc(64 * 32 * 2);
  __bf16*  w2b  = (__bf16*)alloc(64 * 64 * 2);
  __bf16*  w3b  = (__bf16*)alloc(64 * 128 * 2);
  __bf16*  w4b  = (__bf16*)alloc(64 * 64 * 2);
  __bf16*  w5b  = (__bf16*)alloc(64 * 128 * 2);
  __bf16*  w6b  = (__bf16*)alloc(1024 * 192 * 2);
  float*   part = (float*)alloc((size_t)64 * 2048 * 4);     // stat partials
  float*   stat = (float*)alloc(2 * 1024 * 4);
  float*   ssb  = (float*)alloc(2 * 1024 * 4);
  float*   gbuf = (float*)alloc((size_t)B * 1024 * 4);
  float*   tbuf = (float*)alloc((size_t)B * 512 * 4);
  float*   y6   = dist;  // N*N*4 == 16384*1024*4

  // ---- prep
  transpose_x_kernel<<<(B * N * 3 + 255) / 256, 256, 0, stream>>>(x, p, B, N);
  wconv_kernel<<<(64 * 32 + 255) / 256, 256, 0, stream>>>(w1, w1b, 64, 6, 32);
  wconv_kernel<<<(64 * 64 + 255) / 256, 256, 0, stream>>>(w2, w2b, 64, 64, 64);
  wconv_kernel<<<(64 * 128 + 255) / 256, 256, 0, stream>>>(w3, w3b, 64, 128, 128);
  wconv_kernel<<<(64 * 64 + 255) / 256, 256, 0, stream>>>(w4, w4b, 64, 64, 64);
  wconv_kernel<<<(64 * 128 + 255) / 256, 256, 0, stream>>>(w5, w5b, 64, 128, 128);
  wconv_kernel<<<(1024 * 192 + 255) / 256, 256, 0, stream>>>(w6, w6b, 1024, 192, 192);

  auto bn = [&](const float* ybn, long long rows, int C, int S, const float* gg,
                const float* bb) {
    bn_stats_slice_kernel<<<(C * S + 255) / 256, 256, 0, stream>>>(ybn, rows, C, S, part);
    bn_stats_reduce_kernel<<<(2 * C + 255) / 256, 256, 0, stream>>>(part, S, C, stat);
    bn_finalize_kernel<<<(C + 255) / 256, 256, 0, stream>>>(stat, gg, bb, ssb, C,
                                                            1.0f / (float)rows);
  };

  // ---- stage 1: knn on xyz (C=3), edge conv 6->64, conv 64->64, max_k -> x1
  for (int b = 0; b < B; ++b) {
    dist3_kernel<<<((size_t)N * N + 255) / 256, 256, 0, stream>>>(p + (size_t)b * N * 3,
                                                                  dist, N);
    topk_kernel<<<N / 8, 256, 0, stream>>>(dist, nbr + (size_t)b * N * Kn, N);
  }
  edgeconv_wmma_kernel<<<(int)(R / 128), 256, 0, stream>>>(p, nbr, w1b, ybuf, 3, 32,
                                                           N, Kn, 64);
  bn(ybuf, R, 64, 512, g1, b1);
  bn_act_bf16_kernel<<<(int)((R * 64 + 255) / 256), 256, 0, stream>>>(ybuf, ssb, hbf,
                                                                      R * 64, 64);
  gemm_wmma_kernel<<<dim3((int)(R / 128), 1), 256, 0, stream>>>(hbf, w2b, ybuf, 64, 64);
  bn(ybuf, R, 64, 512, g2, b2);
  bn_act_maxk_kernel<<<(int)((NR * 64 + 255) / 256), 256, 0, stream>>>(
      ybuf, ssb, x1f, x1b, catb, 192, 0, NR, Kn, 64);

  // ---- stage 2: knn on x1 (C=64), edge conv 128->64, conv 64->64, max_k -> x2
  rownorm_kernel<<<(int)((NR + 255) / 256), 256, 0, stream>>>(x1f, xx, (int)NR, 64);
  for (int b = 0; b < B; ++b) {
    knn_dist_wmma_kernel<<<(N / 16) * (N / 64) / 8, 256, 0, stream>>>(
        x1b + (size_t)b * N * 64, xx + (size_t)b * N, dist, N, 64);
    topk_kernel<<<N / 8, 256, 0, stream>>>(dist, nbr + (size_t)b * N * Kn, N);
  }
  edgeconv_wmma_kernel<<<(int)(R / 128), 256, 0, stream>>>(x1f, nbr, w3b, ybuf, 64,
                                                           128, N, Kn, 64);
  bn(ybuf, R, 64, 512, g3, b3);
  bn_act_bf16_kernel<<<(int)((R * 64 + 255) / 256), 256, 0, stream>>>(ybuf, ssb, hbf,
                                                                      R * 64, 64);
  gemm_wmma_kernel<<<dim3((int)(R / 128), 1), 256, 0, stream>>>(hbf, w4b, ybuf, 64, 64);
  bn(ybuf, R, 64, 512, g4, b4);
  bn_act_maxk_kernel<<<(int)((NR * 64 + 255) / 256), 256, 0, stream>>>(
      ybuf, ssb, x2f, x2b, catb, 192, 64, NR, Kn, 64);

  // ---- stage 3: knn on x2 (C=64), edge conv 128->64, max_k -> x3 (into cat)
  rownorm_kernel<<<(int)((NR + 255) / 256), 256, 0, stream>>>(x2f, xx, (int)NR, 64);
  for (int b = 0; b < B; ++b) {
    knn_dist_wmma_kernel<<<(N / 16) * (N / 64) / 8, 256, 0, stream>>>(
        x2b + (size_t)b * N * 64, xx + (size_t)b * N, dist, N, 64);
    topk_kernel<<<N / 8, 256, 0, stream>>>(dist, nbr + (size_t)b * N * Kn, N);
  }
  edgeconv_wmma_kernel<<<(int)(R / 128), 256, 0, stream>>>(x2f, nbr, w5b, ybuf, 64,
                                                           128, N, Kn, 64);
  bn(ybuf, R, 64, 512, g5, b5);
  bn_act_maxk_kernel<<<(int)((NR * 64 + 255) / 256), 256, 0, stream>>>(
      ybuf, ssb, x1f /*scratch*/, x1b /*scratch*/, catb, 192, 128, NR, Kn, 64);

  // ---- cat GEMM 192->1024, BN+LReLU, global max pool, final linear + BN
  gemm_wmma_kernel<<<dim3((int)(NR / 128), 16), 256, 0, stream>>>(catb, w6b, y6, 192,
                                                                  1024);
  bn(y6, NR, 1024, 64, g6, b6);
  global_maxn_kernel<<<B, 1024, 0, stream>>>(y6, ssb, gbuf, N, 1024);
  final_linear_kernel<<<(B * 512 + 255) / 256, 256, 0, stream>>>(gbuf, wl, tbuf, B,
                                                                 1024, 512);
  final_bn_kernel<<<(512 + 255) / 256, 256, 0, stream>>>(tbuf, g7, b7, (float*)d_out,
                                                         B, 512);
}
